// MultiGCN_progation_63084479644124
// MI455X (gfx1250) — matrix operand order repaired and checked by reading, not compiled, not run on hardware
//
#include <hip/hip_runtime.h>
#include <stdint.h>

// ---------------------------------------------------------------------------
// Types for CDNA5 WMMA (wave32) + TDM descriptors
// ---------------------------------------------------------------------------
typedef __attribute__((ext_vector_type(16))) __bf16 v16bf;
typedef __attribute__((ext_vector_type(8)))  __bf16 v8bf;
typedef __attribute__((ext_vector_type(8)))  float  v8f;
typedef __attribute__((ext_vector_type(4)))  unsigned int u32x4;
typedef __attribute__((ext_vector_type(8)))  int i32x8;
typedef __attribute__((ext_vector_type(4)))  int i32x4;

#define NN        4096
#define DD        640
#define HH        1000
#define HP        1024
#define WM        4104     // 4096 + 5 RHS + 3 pad
#define WTOT      4101     // 4096 + 5 RHS (cols that matter in LU)
#define KKNN      819
#define NS        40
#define LUBS      64
#define KCH       128      // K-chunk staged in LDS by the TDM
#define EPSF      2.220446049250313e-16f

#define WMMA_BF16(a, b, c) \
  __builtin_amdgcn_wmma_f32_16x16x32_bf16(false, (a), false, (b), (short)0, (c), false, false)

// Tensor Data Mover availability / arity (ROCm7.2 clang-22: 5 args; clang-23: 6 args)
#if defined(__has_builtin)
#  if __has_builtin(__builtin_amdgcn_tensor_load_to_lds)
#    define HAVE_TDM 1
#  endif
#endif
#ifndef HAVE_TDM
#  define HAVE_TDM 0
#endif
#if HAVE_TDM
#  if __clang_major__ >= 23
#    define TDM_CALL(g0, g1, z4, z8) __builtin_amdgcn_tensor_load_to_lds((g0), (g1), (z4), (z4), (z8), 0)
#  else
#    define TDM_CALL(g0, g1, z4, z8) __builtin_amdgcn_tensor_load_to_lds((g0), (g1), (z4), (z4), 0)
#  endif
#endif

// ---------------------------------------------------------------------------
// WMMA fragment helpers (layouts per CDNA5 ISA 7.12.2)
// ---------------------------------------------------------------------------
__device__ __forceinline__ v16bf cat16(v8bf lo, v8bf hi) {
  v16bf r;
#pragma unroll
  for (int e = 0; e < 8; ++e) { r[e] = lo[e]; r[e + 8] = hi[e]; }
  return r;
}

// A (16x32, row-major source): lane<16 -> M=lane, K {0..7,16..23}; lane>=16 -> K {8..15,24..31}
__device__ __forceinline__ v16bf load_a_frag(const __bf16* A, int lda, int m0, int k0, int lane) {
  int m  = m0 + (lane & 15);
  int kb = k0 + ((lane >> 4) << 3);
  const __bf16* p = A + (size_t)m * lda + kb;
  v8bf lo = *(const v8bf*)p;
  v8bf hi = *(const v8bf*)(p + 16);
  return cat16(lo, hi);
}

// B (32x16) fragment for C = A * B^T where B is row-major [N][K]:
// lane<16 -> N=lane, K {0..7,16..23}; lane>=16 -> K {8..15,24..31}
__device__ __forceinline__ v16bf load_bt_frag(const __bf16* B, int ldb, int n0, int k0, int lane) {
  int n  = n0 + (lane & 15);
  int kb = k0 + ((lane >> 4) << 3);
  const __bf16* p = B + (size_t)n * ldb + kb;
  v8bf lo = *(const v8bf*)p;
  v8bf hi = *(const v8bf*)(p + 16);
  return cat16(lo, hi);
}

// C/D 16x16 f32: VGPR v -> M = v + 8*(lane>=16), N = lane&15
__device__ __forceinline__ void store_c(float* C, int ldc, int m0, int n0, int lane, v8f acc) {
  int col   = n0 + (lane & 15);
  int rbase = m0 + ((lane >> 4) << 3);
#pragma unroll
  for (int v = 0; v < 8; ++v) C[(size_t)(rbase + v) * ldc + col] = acc[v];
}

#if HAVE_TDM
// Issue a TDM load of a rows_tile x cols_tile bf16 tile (row-major, stride = tensor_w)
// into LDS at byte offset lds_off. D# layout per CDNA5 ISA ch.8.
__device__ __forceinline__ void tdm_load_tile_bf16(const __bf16* gsrc, unsigned lds_off,
                                                   unsigned rows_tile, unsigned cols_tile,
                                                   unsigned tensor_w, unsigned tensor_h) {
  unsigned long long ga = (unsigned long long)(size_t)gsrc;
  u32x4 g0;
  g0[0] = 1u;                                                   // count=1, user descriptor
  g0[1] = lds_off;                                              // lds_addr (bytes)
  g0[2] = (unsigned)(ga & 0xffffffffu);                         // global_addr[31:0]
  g0[3] = (unsigned)((ga >> 32) & 0x01ffffffu) | (2u << 30);    // global_addr[56:32] | type=2
  i32x8 g1;
  g1[0] = (int)(1u << 16);                                      // data_size=1 (2 bytes)
  g1[1] = (int)((tensor_w & 0xffffu) << 16);                    // tensor_dim0 lo16
  g1[2] = (int)((tensor_w >> 16) | ((tensor_h & 0xffffu) << 16)); // dim0 hi16 | dim1 lo16
  g1[3] = (int)((tensor_h >> 16) | (cols_tile << 16));          // dim1 hi16 | tile_dim0
  g1[4] = (int)rows_tile;                                       // tile_dim1 (tile_dim2 = 0)
  g1[5] = (int)tensor_w;                                        // tensor_dim0_stride lo32
  g1[6] = 0;
  g1[7] = 0;
  i32x4 z4 = {0, 0, 0, 0};
  i32x8 z8 = {0, 0, 0, 0, 0, 0, 0, 0};
  (void)z8;
  TDM_CALL(g0, g1, z4, z8);
}
#endif

// ---------------------------------------------------------------------------
// NT GEMM: C[M,N] = A[M,K] * B[N,K]^T, bf16 in / f32 out.
// Grid: (N/256, M/32). Block = 256 threads = 8 waves; wave w owns the 32x32
// tile at (m0 = 32*blockIdx.y, n0 = 32*(8*blockIdx.x + w)).
// The 32-row A-panel is shared by all 8 waves: staged into LDS in 32xKCH
// chunks by the Tensor Data Mover (double-buffered, TENSORcnt + barrier),
// while per-wave B fragments stream from global as b128 loads.
// ---------------------------------------------------------------------------
__global__ void gemm_nt_bf16(const __bf16* A, const __bf16* B, float* C,
                             int M, int N, int K, int lda, int ldb, int ldc) {
  __shared__ __bf16 As[2][32 * KCH];
  int lane = threadIdx.x & 31;
  int wave = threadIdx.x >> 5;
  int m0 = blockIdx.y << 5;
  int n0 = ((blockIdx.x << 3) + wave) << 5;

  const __bf16* pb = B + (size_t)(n0 + lane) * ldb;   // one B row per lane (prefetch)
  int nch = K / KCH;

  // stage chunk 0
#if HAVE_TDM
  if (wave == 0) {
    tdm_load_tile_bf16(A + (size_t)m0 * lda, (unsigned)(size_t)&As[0][0],
                       32u, (unsigned)KCH, (unsigned)lda, (unsigned)M);
    __builtin_amdgcn_s_wait_tensorcnt(0);
  }
#else
  for (int i = threadIdx.x; i < (32 * KCH) / 8; i += 256) {
    int r = (i << 3) >> 7, c = (i << 3) & (KCH - 1);
    *(v8bf*)&As[0][r * KCH + c] = *(const v8bf*)(A + (size_t)(m0 + r) * lda + c);
  }
#endif
  __syncthreads();

  v8f acc00 = {}, acc01 = {}, acc10 = {}, acc11 = {};
  for (int ch = 0; ch < nch; ++ch) {
    int kc = ch * KCH;
    int buf = ch & 1;
#if HAVE_TDM
    if (wave == 0 && ch + 1 < nch) {
      tdm_load_tile_bf16(A + (size_t)m0 * lda + (kc + KCH), (unsigned)(size_t)&As[buf ^ 1][0],
                         32u, (unsigned)KCH, (unsigned)lda, (unsigned)M);
    }
#else
    if (ch + 1 < nch) {
      for (int i = threadIdx.x; i < (32 * KCH) / 8; i += 256) {
        int r = (i << 3) >> 7, c = (i << 3) & (KCH - 1);
        *(v8bf*)&As[buf ^ 1][r * KCH + c] =
            *(const v8bf*)(A + (size_t)(m0 + r) * lda + (kc + KCH) + c);
      }
    }
#endif
    if (kc + KCH < K)
      __builtin_prefetch((const void*)(pb + kc + KCH), 0, 3);

    const __bf16* Ash = &As[buf][0];
#pragma unroll
    for (int kk = 0; kk < KCH; kk += 32) {
      v16bf a0 = load_a_frag(Ash, KCH, 0,  kk, lane);
      v16bf a1 = load_a_frag(Ash, KCH, 16, kk, lane);
      v16bf b0 = load_bt_frag(B, ldb, n0,      kc + kk, lane);
      v16bf b1 = load_bt_frag(B, ldb, n0 + 16, kc + kk, lane);
      acc00 = WMMA_BF16(a0, b0, acc00);
      acc01 = WMMA_BF16(a0, b1, acc01);
      acc10 = WMMA_BF16(a1, b0, acc10);
      acc11 = WMMA_BF16(a1, b1, acc11);
    }
#if HAVE_TDM
    if (wave == 0 && ch + 1 < nch) __builtin_amdgcn_s_wait_tensorcnt(0);
#endif
    __syncthreads();
  }
  store_c(C, ldc, m0,      n0,      lane, acc00);
  store_c(C, ldc, m0,      n0 + 16, lane, acc01);
  store_c(C, ldc, m0 + 16, n0,      lane, acc10);
  store_c(C, ldc, m0 + 16, n0 + 16, lane, acc11);
}

// ---------------------------------------------------------------------------
// LDS-tiled bf16 transpose: out[c][r] = in[r][c]; rows, cols multiples of 32.
// ---------------------------------------------------------------------------
__global__ void transpose_bf16(const __bf16* in, __bf16* out, int rows, int cols) {
  __shared__ __bf16 t[32][33];
  int x0 = blockIdx.x * 32, y0 = blockIdx.y * 32;
  for (int dy = threadIdx.y; dy < 32; dy += 8)
    t[dy][threadIdx.x] = in[(size_t)(y0 + dy) * cols + x0 + threadIdx.x];
  __syncthreads();
  for (int dy = threadIdx.y; dy < 32; dy += 8)
    out[(size_t)(x0 + dy) * rows + y0 + threadIdx.x] = t[threadIdx.x][dy];
}

// ---------------------------------------------------------------------------
// Elementwise / reduction helpers
// ---------------------------------------------------------------------------
__global__ void cvt_pad_bf16(const float* src, int scols, __bf16* dst, int dcols, int rows) {
  size_t total = (size_t)rows * dcols;
  for (size_t i = (size_t)blockIdx.x * blockDim.x + threadIdx.x; i < total;
       i += (size_t)gridDim.x * blockDim.x) {
    int c = (int)(i % dcols);
    int r = (int)(i / dcols);
    float v = (c < scols) ? src[(size_t)r * scols + c] : 0.0f;
    dst[i] = (__bf16)v;
  }
}

__global__ void rowsq_kernel(const float* X, int ld, int cols, float* out) {
  int row = blockIdx.x;
  const float* x = X + (size_t)row * ld;
  float s = 0.0f;
  for (int j = threadIdx.x; j < cols; j += blockDim.x) { float v = x[j]; s += v * v; }
  __shared__ float red[256];
  red[threadIdx.x] = s; __syncthreads();
  for (int st = blockDim.x >> 1; st > 0; st >>= 1) {
    if (threadIdx.x < (unsigned)st) red[threadIdx.x] += red[threadIdx.x + st];
    __syncthreads();
  }
  if (threadIdx.x == 0) out[row] = red[0];
}

__global__ void dist_exp(float* G, const float* sq, int ld, int n, float inv_s2) {
  size_t total = (size_t)n * ld;
  for (size_t i = (size_t)blockIdx.x * blockDim.x + threadIdx.x; i < total;
       i += (size_t)gridDim.x * blockDim.x) {
    int r = (int)(i / ld), c = (int)(i % ld);
    float d2 = fmaxf(sq[r] + sq[c] - 2.0f * G[i], 0.0f);
    G[i] = expf(-d2 * inv_s2);
  }
}

// per-row k-th largest via bisection on (0, 1]
__global__ void topk_thresh(const float* Ae, int ld, int n, int k, float* thr) {
  int row = blockIdx.x;
  const float* a = Ae + (size_t)row * ld;
  __shared__ int s_cnt;
  float lo = 0.0f, hi = 1.0f + 1e-6f;
  for (int it = 0; it < 30; ++it) {
    float mid = 0.5f * (lo + hi);
    if (threadIdx.x == 0) s_cnt = 0;
    __syncthreads();
    int c = 0;
    for (int j = threadIdx.x; j < n; j += blockDim.x) c += (a[j] > mid) ? 1 : 0;
    atomicAdd(&s_cnt, c);
    __syncthreads();
    int cnt = s_cnt;
    __syncthreads();
    if (cnt >= k) lo = mid; else hi = mid;
  }
  if (threadIdx.x == 0) thr[row] = 0.5f * (lo + hi);
}

// A := eye + mask*A_e (mask = topk excl self); optional rowsum
__global__ void apply_knn(float* A, const float* thr, int ld, int n, float* rowsum) {
  int row = blockIdx.x;
  float t = thr[row];
  float* a = A + (size_t)row * ld;
  float s = 0.0f;
  for (int j = threadIdx.x; j < n; j += blockDim.x) {
    float v = a[j];
    float o = (j == row) ? 1.0f : ((v >= t) ? v : 0.0f);
    a[j] = o;
    s += o;
  }
  if (rowsum) {
    __shared__ float red[256];
    red[threadIdx.x] = s; __syncthreads();
    for (int st = blockDim.x >> 1; st > 0; st >>= 1) {
      if (threadIdx.x < (unsigned)st) red[threadIdx.x] += red[threadIdx.x + st];
      __syncthreads();
    }
    if (threadIdx.x == 0) rowsum[row] = red[0];
  }
}

__global__ void symnorm_bf16(float* A, const float* d, int ld, int n, __bf16* Abf) {
  size_t total = (size_t)n * ld;
  for (size_t i = (size_t)blockIdx.x * blockDim.x + threadIdx.x; i < total;
       i += (size_t)gridDim.x * blockDim.x) {
    int r = (int)(i / ld), c = (int)(i % ld);
    float v = A[i] * rsqrtf(d[r] * d[c]);
    A[i] = v;
    Abf[i] = (__bf16)v;
  }
}

__global__ void softmax3(const float* a, const float* b, const float* c, float* out) {
  float x0 = a[0], x1 = b[0], x2 = c[0];
  float m = fmaxf(x0, fmaxf(x1, x2));
  float e0 = expf(x0 - m), e1 = expf(x1 - m), e2 = expf(x2 - m);
  float s = e0 + e1 + e2;
  out[0] = e0 / s; out[1] = e1 / s; out[2] = e2 / s;
}

// Abig = aifa0*I + aifa1*An + aifa2*An2  (bf16 out)
__global__ void combine_abig(const float* An, const float* An2, const float* aifa,
                             int n, __bf16* out) {
  float a0 = aifa[0], a1 = aifa[1], a2 = aifa[2];
  size_t total = (size_t)n * n;
  for (size_t i = (size_t)blockIdx.x * blockDim.x + threadIdx.x; i < total;
       i += (size_t)gridDim.x * blockDim.x) {
    int r = (int)(i / n), c = (int)(i % n);
    float v = a1 * An[i] + a2 * An2[i] + ((r == c) ? a0 : 0.0f);
    out[i] = (__bf16)v;
  }
}

__global__ void pad_vec(const float* b, int n, float* out, int npad) {
  int i = blockIdx.x * blockDim.x + threadIdx.x;
  if (i < npad) out[i] = (i < n) ? b[i] : 0.0f;
}

__global__ void add_bias(float* X, const float* bias, int rows, int ld) {
  size_t total = (size_t)rows * ld;
  for (size_t i = (size_t)blockIdx.x * blockDim.x + threadIdx.x; i < total;
       i += (size_t)gridDim.x * blockDim.x)
    X[i] += bias[i % ld];
}

// M = eye - alpha*S + EPS, with y (one-hot labels) appended in cols N..N+4, pad cols 0.
__global__ void build_M(float* Mt, const float* S, const float* alphaPtr,
                        const int* s_label, int n, int wm) {
  float alpha = alphaPtr[0];
  size_t total = (size_t)n * wm;
  for (size_t i = (size_t)blockIdx.x * blockDim.x + threadIdx.x; i < total;
       i += (size_t)gridDim.x * blockDim.x) {
    int r = (int)(i / wm), c = (int)(i % wm);
    float v;
    if (c < n) {
      v = ((r == c) ? 1.0f : 0.0f) - alpha * S[(size_t)r * n + c] + EPSF;
    } else {
      int cc = c - n;
      v = (cc < 5 && r < NS && s_label[r] == cc) ? 1.0f : 0.0f;
    }
    Mt[i] = v;
  }
}

// ---------------------------------------------------------------------------
// Blocked LU (no pivoting), BS = 64, matrix width WM (RHS appended)
// ---------------------------------------------------------------------------
__global__ void lu_diag(float* Mt, int wm, int p) {
  __shared__ float D[LUBS][LUBS + 1];
  for (int i = threadIdx.x; i < LUBS * LUBS; i += blockDim.x) {
    int r = i >> 6, c = i & 63;
    D[r][c] = Mt[(size_t)(p + r) * wm + (p + c)];
  }
  __syncthreads();
  for (int j = 0; j < LUBS - 1; ++j) {
    int i = threadIdx.x;
    if (i > j && i < LUBS) {
      float l = D[i][j] / D[j][j];
      D[i][j] = l;
      for (int c = j + 1; c < LUBS; ++c) D[i][c] -= l * D[j][c];
    }
    __syncthreads();
  }
  for (int i = threadIdx.x; i < LUBS * LUBS; i += blockDim.x) {
    int r = i >> 6, c = i & 63;
    Mt[(size_t)(p + r) * wm + (p + c)] = D[r][c];
  }
}

// U12 = L11^-1 * A12 : one column per thread (forward substitution)
__global__ void lu_u12(float* Mt, int wm, int p, int wtot) {
  __shared__ float L[LUBS][LUBS + 1];
  __shared__ float U[LUBS][LUBS + 1];   // per-thread column U[*][tid]
  int c = p + LUBS + blockIdx.x * LUBS + threadIdx.x;
  for (int i = threadIdx.x; i < LUBS * LUBS; i += blockDim.x) {
    int r = i >> 6, cc = i & 63;
    L[r][cc] = Mt[(size_t)(p + r) * wm + (p + cc)];
  }
  __syncthreads();
  bool act = c < wtot;
  for (int t = 0; t < LUBS; ++t) {
    float u = act ? Mt[(size_t)(p + t) * wm + c] : 0.0f;
    for (int s = 0; s < t; ++s) u -= L[t][s] * U[s][threadIdx.x];
    U[t][threadIdx.x] = u;
  }
  if (act)
    for (int t = 0; t < LUBS; ++t) Mt[(size_t)(p + t) * wm + c] = U[t][threadIdx.x];
}

// L21 = A21 * U11^-1 : one row per thread
__global__ void lu_l21(float* Mt, int wm, int p, int n) {
  __shared__ float U[LUBS][LUBS + 1];
  __shared__ float Lb[LUBS][LUBS + 1];  // per-thread column Lb[*][tid]
  int r = p + LUBS + blockIdx.x * LUBS + threadIdx.x;
  for (int i = threadIdx.x; i < LUBS * LUBS; i += blockDim.x) {
    int rr = i >> 6, cc = i & 63;
    U[rr][cc] = Mt[(size_t)(p + rr) * wm + (p + cc)];
  }
  __syncthreads();
  bool act = r < n;
  for (int j = 0; j < LUBS; ++j) {
    float l = act ? Mt[(size_t)r * wm + (p + j)] : 0.0f;
    for (int s = 0; s < j; ++s) l -= Lb[s][threadIdx.x] * U[s][j];
    l /= U[j][j];
    Lb[j][threadIdx.x] = l;
  }
  if (act)
    for (int j = 0; j < LUBS; ++j) Mt[(size_t)r * wm + (p + j)] = Lb[j][threadIdx.x];
}

// A22 -= L21 * U12 : f32 LDS-tiled 64x64 tile per block, 4x4 micro-tile/thread
__global__ void lu_trail(float* Mt, int wm, int p, int n, int wtot) {
  __shared__ float Ls[LUBS][LUBS + 1];
  __shared__ float Us[LUBS][LUBS + 1];
  int r0 = p + LUBS + blockIdx.y * LUBS;
  int c0 = p + LUBS + blockIdx.x * LUBS;
  int tid = threadIdx.y * 16 + threadIdx.x;
  for (int i = tid; i < LUBS * LUBS; i += 256) {
    int rr = i >> 6, cc = i & 63;
    int gr = r0 + rr;
    int gc = c0 + cc;
    Ls[rr][cc] = (gr < n)    ? Mt[(size_t)gr * wm + (p + cc)] : 0.0f;
    Us[rr][cc] = (gc < wtot) ? Mt[(size_t)(p + rr) * wm + gc] : 0.0f;
  }
  __syncthreads();
  float acc[4][4] = {};
  int ri = threadIdx.y * 4, ci = threadIdx.x * 4;
  for (int t = 0; t < LUBS; ++t) {
    float a0 = Ls[ri + 0][t], a1 = Ls[ri + 1][t], a2 = Ls[ri + 2][t], a3 = Ls[ri + 3][t];
    float b0 = Us[t][ci + 0], b1 = Us[t][ci + 1], b2 = Us[t][ci + 2], b3 = Us[t][ci + 3];
    acc[0][0] += a0 * b0; acc[0][1] += a0 * b1; acc[0][2] += a0 * b2; acc[0][3] += a0 * b3;
    acc[1][0] += a1 * b0; acc[1][1] += a1 * b1; acc[1][2] += a1 * b2; acc[1][3] += a1 * b3;
    acc[2][0] += a2 * b0; acc[2][1] += a2 * b1; acc[2][2] += a2 * b2; acc[2][3] += a2 * b3;
    acc[3][0] += a3 * b0; acc[3][1] += a3 * b1; acc[3][2] += a3 * b2; acc[3][3] += a3 * b3;
  }
#pragma unroll
  for (int i = 0; i < 4; ++i)
#pragma unroll
    for (int j = 0; j < 4; ++j) {
      int gr = r0 + ri + i, gc = c0 + ci + j;
      if (gr < n && gc < wtot) Mt[(size_t)gr * wm + gc] -= acc[i][j];
    }
}

// Column-sweep back substitution, 5 RHS in cols N..N+4 of Mt
__global__ void backsub(float* Mt, int n, int wm, float* sol) {
  __shared__ float xi[5];
  for (int i = n - 1; i >= 0; --i) {
    if (threadIdx.x < 5) {
      float v = Mt[(size_t)i * wm + n + threadIdx.x] / Mt[(size_t)i * wm + i];
      xi[threadIdx.x] = v;
      sol[i * 5 + threadIdx.x] = v;
    }
    __syncthreads();
    float x0 = xi[0], x1 = xi[1], x2 = xi[2], x3 = xi[3], x4 = xi[4];
    for (int r = threadIdx.x; r < i; r += blockDim.x) {
      float u = Mt[(size_t)r * wm + i];
      float* yr = Mt + (size_t)r * wm + n;
      yr[0] -= u * x0; yr[1] -= u * x1; yr[2] -= u * x2; yr[3] -= u * x3; yr[4] -= u * x4;
    }
    __syncthreads();
  }
}

__global__ void write_out(const float* sol, float* out, int n, int n_s) {
  int total = n * 5 + (n - n_s) * 5;
  for (int i = blockIdx.x * blockDim.x + threadIdx.x; i < total; i += gridDim.x * blockDim.x) {
    if (i < n * 5) out[i] = sol[i];
    else           out[i] = sol[n_s * 5 + (i - n * 5)];
  }
}

// ---------------------------------------------------------------------------
// Host orchestration
// ---------------------------------------------------------------------------
extern "C" void kernel_launch(void* const* d_in, const int* in_sizes, int n_in,
                              void* d_out, int out_size, void* d_ws, size_t ws_size,
                              hipStream_t stream) {
  const float* features = (const float*)d_in[0];
  const int*   s_label  = (const int*)d_in[1];
  const float* Wg       = (const float*)d_in[2];
  const float* bg       = (const float*)d_in[3];
  const float* aifa1    = (const float*)d_in[4];
  const float* aifa2    = (const float*)d_in[5];
  const float* aifa3    = (const float*)d_in[6];
  const float* alpha    = (const float*)d_in[7];
  float* out = (float*)d_out;
  (void)in_sizes; (void)n_in; (void)out_size; (void)ws_size;

  uint8_t* w = (uint8_t*)d_ws;
  size_t off = 0;
  auto take = [&](size_t bytes) -> void* {
    void* p = w + off;
    off += (bytes + 255) & ~(size_t)255;
    return p;
  };

  float*  G    = (float*) take((size_t)NN * NN * 4);   // Gram / A_e / An / S
  float*  MT   = (float*) take((size_t)NN * WM * 4);   // An2 (ld=NN), later LU matrix (ld=WM)
  __bf16* AB   = (__bf16*)take((size_t)NN * NN * 2);   // An bf16 -> Abig bf16
  __bf16* ABT  = (__bf16*)take((size_t)NN * NN * 2);   // An^T bf16
  float*  XF   = (float*) take((size_t)NN * HP * 4);   // XW f32 -> x f32
  __bf16* XB   = (__bf16*)take((size_t)NN * HP * 2);   // XW bf16 -> x bf16
  __bf16* XBT  = (__bf16*)take((size_t)NN * HP * 2);   // XW^T bf16
  __bf16* FB   = (__bf16*)take((size_t)NN * DD * 2);
  __bf16* WB   = (__bf16*)take((size_t)DD * HP * 2);   // W padded [640 x 1024]
  __bf16* WT   = (__bf16*)take((size_t)HP * DD * 2);   // W^T [1024 x 640]
  float*  sq   = (float*) take(NN * 4);
  float*  thr  = (float*) take(NN * 4);
  float*  dv   = (float*) take(NN * 4);
  float*  aifa = (float*) take(4 * 4);
  float*  bp   = (float*) take(HP * 4);
  float*  sol  = (float*) take((size_t)NN * 5 * 4);

  const int T = 256;
  long nnTot = (long)NN * NN;
  int nnBlk  = (int)((nnTot + T - 1) / T);
  dim3 tb(32, 8);

  // 1) bf16 copies of features, padded W, and W^T
  cvt_pad_bf16<<<(int)(((size_t)NN * DD + T - 1) / T), T, 0, stream>>>(features, DD, FB, DD, NN);
  cvt_pad_bf16<<<(int)(((size_t)DD * HP + T - 1) / T), T, 0, stream>>>(Wg, HH, WB, HP, DD);
  transpose_bf16<<<dim3(HP / 32, DD / 32), tb, 0, stream>>>(WB, WT, DD, HP);

  // 2) first kNN adjacency: G = F F^T, exp kernel, topk, mask, sym-normalize
  rowsq_kernel<<<NN, T, 0, stream>>>(features, DD, DD, sq);
  gemm_nt_bf16<<<dim3(NN / 256, NN / 32), T, 0, stream>>>(FB, FB, G, NN, NN, DD, DD, DD, NN);
  dist_exp<<<nnBlk, T, 0, stream>>>(G, sq, NN, NN, 1.0f / 9.0f);
  topk_thresh<<<NN, T, 0, stream>>>(G, NN, NN, KKNN, thr);
  apply_knn<<<NN, T, 0, stream>>>(G, thr, NN, NN, dv);
  symnorm_bf16<<<nnBlk, T, 0, stream>>>(G, dv, NN, NN, AB);   // G = An f32, AB = An bf16

  // 3) An2 = An @ An  ==  An * (An^T)^T  (all-NT WMMA path)
  transpose_bf16<<<dim3(NN / 32, NN / 32), tb, 0, stream>>>(AB, ABT, NN, NN);
  gemm_nt_bf16<<<dim3(NN / 256, NN / 32), T, 0, stream>>>(AB, ABT, MT, NN, NN, NN, NN, NN, NN);

  // 4) Abig = softmax(aifa) . {I, An, An2} as bf16 (overwrites AB)
  softmax3<<<1, 1, 0, stream>>>(aifa1, aifa2, aifa3, aifa);
  combine_abig<<<nnBlk, T, 0, stream>>>(G, MT, aifa, NN, AB);

  // 5) x = Abig @ (F W) + b   (F@W and Abig@XW both as NT via transposes)
  gemm_nt_bf16<<<dim3(HP / 256, NN / 32), T, 0, stream>>>(FB, WT, XF, NN, HP, DD, DD, DD, HP);
  cvt_pad_bf16<<<(int)(((size_t)NN * HP + T - 1) / T), T, 0, stream>>>(XF, HP, XB, HP, NN);
  transpose_bf16<<<dim3(HP / 32, NN / 32), tb, 0, stream>>>(XB, XBT, NN, HP);
  gemm_nt_bf16<<<dim3(HP / 256, NN / 32), T, 0, stream>>>(AB, XBT, XF, NN, HP, NN, NN, NN, HP);
  pad_vec<<<(HP + T - 1) / T, T, 0, stream>>>(bg, HH, bp, HP);
  add_bias<<<(int)(((size_t)NN * HP + T - 1) / T), T, 0, stream>>>(XF, bp, NN, HP);

  // 6) second kNN adjacency on x: S = eye + mask * exp(-d2/30)
  rowsq_kernel<<<NN, T, 0, stream>>>(XF, HP, HP, sq);
  cvt_pad_bf16<<<(int)(((size_t)NN * HP + T - 1) / T), T, 0, stream>>>(XF, HP, XB, HP, NN);
  gemm_nt_bf16<<<dim3(NN / 256, NN / 32), T, 0, stream>>>(XB, XB, G, NN, NN, HP, HP, HP, NN);
  dist_exp<<<nnBlk, T, 0, stream>>>(G, sq, NN, NN, 1.0f / 30.0f);
  topk_thresh<<<NN, T, 0, stream>>>(G, NN, NN, KKNN, thr);
  apply_knn<<<NN, T, 0, stream>>>(G, thr, NN, NN, nullptr);   // G = S

  // 7) M = I - alpha*S + EPS with y appended; blocked LU (no pivot); solve
  build_M<<<(int)(((size_t)NN * WM + T - 1) / T), T, 0, stream>>>(MT, G, alpha, s_label, NN, WM);

  for (int p = 0; p < NN; p += LUBS) {
    lu_diag<<<1, LUBS, 0, stream>>>(MT, WM, p);
    int cols = WTOT - (p + LUBS);
    int rows = NN - (p + LUBS);
    if (cols > 0)
      lu_u12<<<(cols + LUBS - 1) / LUBS, LUBS, 0, stream>>>(MT, WM, p, WTOT);
    if (rows > 0)
      lu_l21<<<(rows + LUBS - 1) / LUBS, LUBS, 0, stream>>>(MT, WM, p, NN);
    if (rows > 0 && cols > 0) {
      dim3 grid((cols + LUBS - 1) / LUBS, (rows + LUBS - 1) / LUBS);
      lu_trail<<<grid, dim3(16, 16), 0, stream>>>(MT, WM, p, NN, WTOT);
    }
  }
  backsub<<<1, T, 0, stream>>>(MT, NN, WM, sol);

  // 8) outputs: F_all (4096x5) then Fq (4056x5)
  write_out<<<160, T, 0, stream>>>(sol, out, NN, NS);
}
